// GCNWithEdge_65335042506766
// MI455X (gfx1250) — compile-verified, matrix operand order checked by reading, not run on hardware
//
#include <hip/hip_runtime.h>
#include <hip/hip_bf16.h>

typedef _Float16 half_t;
typedef __attribute__((ext_vector_type(16))) _Float16 v16h;
typedef __attribute__((ext_vector_type(8)))  _Float16 v8h;
typedef __attribute__((ext_vector_type(2)))  _Float16 h2;
typedef __attribute__((ext_vector_type(8)))  float    v8f;

#define NNODES 20000
#define NEDGES 320000
#define EAD 8
#define DIN 16
#define DOUT 16
#define HDIM 256          // din*dout
#define WAVES_PER_WG 8
#define BLOCK (WAVES_PER_WG * 32)
#define EDGE_WGS 256
#define SLOPE 0.01f
#define HPAD 264          // h-tile row stride (halves): 132 dwords == 4 mod 64 -> full-bank A loads

__device__ __forceinline__ float leaky_f(float v) { return fmaxf(v, SLOPE * v); }

// B fragment stored as two 16B chunks per lane: [frag][half][lane][8 halves]
// -> per-lane stride 16 B (4 dwords): one ds_load_b128 from 32 lanes covers all 64 banks.
__device__ __forceinline__ v16h load_bfrag(const half_t* fragBase, int lane)
{
    const v8h lo = *(const v8h*)(fragBase + lane * 8);
    const v8h hi = *(const v8h*)(fragBase + 256 + lane * 8);
    return __builtin_shufflevector(lo, hi, 0, 1, 2, 3, 4, 5, 6, 7,
                                   8, 9, 10, 11, 12, 13, 14, 15);
}

// ---------------------------------------------------------------------------
// Fused per-edge NNConv message kernel. Each wave owns 16-edge tiles:
//   h  = leaky(ea~ @ w1~)          ea~ K-padded 8->32 with ea~[8]=1, w1~[8]=b1
//   Wf = h @ w2 + b2               8 chained K=32 WMMAs per N-tile
//   msg[e,o] += x[src[e],nt] * Wf_nt[e,o]   (N-tile nt == input dim i)
//   atomic scatter into agg[dst] (+ cnt on layer 0)
// ---------------------------------------------------------------------------
__launch_bounds__(BLOCK, 1)
__global__ void nnconv_edge_kernel(const float* __restrict__ x,
                                   const int*   __restrict__ edge_index,
                                   const float* __restrict__ ea,
                                   const float* __restrict__ w1,
                                   const float* __restrict__ b1,
                                   const float* __restrict__ w2,
                                   const float* __restrict__ b2,
                                   float* __restrict__ agg,
                                   float* __restrict__ cnt,
                                   int add_cnt)
{
    __shared__ __align__(32) half_t w2B[128 * 512];                  // 128 KB  [nt*8+kk][half][lane][8]
    __shared__ __align__(32) half_t w1B[16 * 512];                   // 16 KB   [nt][half][lane][8]
    __shared__ __align__(32) half_t hT[WAVES_PER_WG][16][HPAD];      // per-wave h tile, row-major
    __shared__ __align__(16) float  xT[WAVES_PER_WG][16][16];        // src feats, TRANSPOSED: [dim][edge]
    __shared__ int dT[WAVES_PER_WG][16];                             // dst node ids

    const int tid = threadIdx.x;

    // ---- pack w2 into bank-optimal B-fragment chunks (f32 -> f16) ----
    for (int i = tid; i < 128 * 512; i += BLOCK) {
        const int fr   = i >> 9;              // nt*8 + kk
        const int rem  = i & 511;
        const int hf   = rem >> 8;            // which 8-half chunk
        const int lane = (rem >> 3) & 31;
        const int j    = (rem & 7) + hf * 8;  // 0..15 within fragment
        const int kk   = fr & 7;
        const int nt   = fr >> 3;
        const int k    = kk * 32 + ((lane >> 4) & 1) * 16 + j;
        const int c    = nt * 16 + (lane & 15);
        w2B[i] = (half_t)w2[k * HDIM + c];
    }
    // ---- pack w1 (K padded 8 -> 32); K==8 row carries b1 (matched by aea[8]=1) ----
    for (int i = tid; i < 16 * 512; i += BLOCK) {
        const int nt   = i >> 9;
        const int rem  = i & 511;
        const int hf   = rem >> 8;
        const int lane = (rem >> 3) & 31;
        const int j    = (rem & 7) + hf * 8;
        const int k    = ((lane >> 4) & 1) * 16 + j;
        const int c    = nt * 16 + (lane & 15);
        float v = 0.f;
        if (k < EAD)       v = w1[k * HDIM + c];
        else if (k == EAD) v = b1[c];
        w1B[i] = (half_t)v;
    }
    __syncthreads();

    const int lane = tid & 31;
    const int wv   = tid >> 5;
    const int m    = lane & 15;    // A/D row or B/D column within fragment
    const int hi   = lane >> 4;    // 0: K 0..15 / rows 0..7 ; 1: K 16..31 / rows 8..15
    const int kb   = hi * 16;

    // per-lane b2 fragment (D-layout column = lane&15)
    float b2v[16];
#pragma unroll
    for (int nt = 0; nt < 16; ++nt) b2v[nt] = b2[nt * 16 + m];

    const h2 slope2 = {(half_t)SLOPE, (half_t)SLOPE};

    const int nTiles     = NEDGES / 16;
    const int waveGlobal = blockIdx.x * WAVES_PER_WG + wv;
    const int nWaves     = gridDim.x * WAVES_PER_WG;

    for (int tile = waveGlobal; tile < nTiles; tile += nWaves) {
        const int e0 = tile * 16;

        // ---- gather: edge_attr (A frag, K-padded + bias lane), src feats, dst ids ----
        v16h aea = {};
        if (hi == 0) {
            const int e = e0 + m;
            const float4* eap = (const float4*)(ea + (size_t)e * EAD);
            const float4 ea0 = eap[0], ea1 = eap[1];
            aea[0] = (half_t)ea0.x; aea[1] = (half_t)ea0.y;
            aea[2] = (half_t)ea0.z; aea[3] = (half_t)ea0.w;
            aea[4] = (half_t)ea1.x; aea[5] = (half_t)ea1.y;
            aea[6] = (half_t)ea1.z; aea[7] = (half_t)ea1.w;
            aea[8] = (half_t)1.f;                    // picks up b1 row of w1B
            const int s = edge_index[e];             // src
            const int d = edge_index[NEDGES + e];    // dst
            dT[wv][m] = d;
            const float* xp = x + (size_t)s * DIN;
#pragma unroll
            for (int j = 0; j < DIN; ++j) xT[wv][j][m] = xp[j];   // transposed store
        }

        // ---- step 1: h = leaky(ea~ @ w1~) -> LDS h tile (packed f16 epilogue) ----
#pragma unroll
        for (int nt = 0; nt < 16; ++nt) {
            v8f d0 = {};
            const v16h bf = load_bfrag(&w1B[nt * 512], lane);
            d0 = __builtin_amdgcn_wmma_f32_16x16x32_f16(false, aea, false, bf,
                                                        (short)0, d0, false, false);
#pragma unroll
            for (int r = 0; r < 8; r += 2) {
                h2 p = __builtin_bit_cast(h2, __builtin_amdgcn_cvt_pkrtz(d0[r], d0[r + 1]));
                p = __builtin_elementwise_max(p, p * slope2);      // leaky in packed f16
                const int row = r + hi * 8;
                hT[wv][row][nt * 16 + m]     = p[0];
                hT[wv][row + 1][nt * 16 + m] = p[1];
            }
        }

        // ---- preload all 8 A fragments of the h tile ----
        v16h af[8];
#pragma unroll
        for (int kk = 0; kk < 8; ++kk) {
            const v8h lo = *(const v8h*)&hT[wv][m][kk * 32 + kb];
            const v8h hh = *(const v8h*)&hT[wv][m][kk * 32 + kb + 8];
            af[kk] = __builtin_shufflevector(lo, hh, 0, 1, 2, 3, 4, 5, 6, 7,
                                             8, 9, 10, 11, 12, 13, 14, 15);
        }

        // ---- step 2+3: per N-tile GEMM vs w2, fold in x[src] scalar ----
        v8f msg = {};
#pragma unroll
        for (int nt = 0; nt < 16; ++nt) {
            v8f acc;
#pragma unroll
            for (int r = 0; r < 8; ++r) acc[r] = b2v[nt];   // bias as accumulator init
#pragma unroll
            for (int kk = 0; kk < 8; ++kk) {
                const v16h bf = load_bfrag(&w2B[(nt * 8 + kk) * 512], lane);
                acc = __builtin_amdgcn_wmma_f32_16x16x32_f16(false, af[kk], false, bf,
                                                             (short)0, acc, false, false);
            }
            // x[src] scalars for this input dim: 8 contiguous floats, LDS broadcast
            const float4* xr = (const float4*)&xT[wv][nt][hi * 8];
            const float4 x0 = xr[0], x1 = xr[1];
            msg[0] += x0.x * acc[0]; msg[1] += x0.y * acc[1];
            msg[2] += x0.z * acc[2]; msg[3] += x0.w * acc[3];
            msg[4] += x1.x * acc[4]; msg[5] += x1.y * acc[5];
            msg[6] += x1.z * acc[6]; msg[7] += x1.w * acc[7];
        }

        // ---- step 4: atomic scatter-mean numerator + degree count ----
#pragma unroll
        for (int r = 0; r < 8; ++r) {
            const int row = r + hi * 8;
            const int d   = dT[wv][row];
            unsafeAtomicAdd(&agg[(size_t)d * DOUT + m], msg[r]);
        }
        if (add_cnt && hi == 0) {
            unsafeAtomicAdd(&cnt[dT[wv][m]], 1.0f);
        }
    }
}

// out[n,o] = agg[n,o]/max(cnt[n],1) + x[n,:]@root[:,o] + bias[o]  (opt. leaky)
__global__ void nnconv_node_kernel(const float* __restrict__ agg,
                                   const float* __restrict__ cnt,
                                   const float* __restrict__ xin,
                                   const float* __restrict__ root,
                                   const float* __restrict__ bias,
                                   float* __restrict__ out,
                                   int do_leaky)
{
    __shared__ float rootS[DIN * DOUT];
    if (threadIdx.x < DIN * DOUT) rootS[threadIdx.x] = root[threadIdx.x];
    __syncthreads();

    const int idx = blockIdx.x * blockDim.x + threadIdx.x;
    if (idx >= NNODES * DOUT) return;
    const int n = idx >> 4;
    const int o = idx & 15;

    const float c    = cnt[n];
    const float mean = agg[idx] / fmaxf(c, 1.0f);
    float acc = bias[o];
    const float* xr = xin + (size_t)n * DIN;
#pragma unroll
    for (int i = 0; i < DIN; ++i) acc += xr[i] * rootS[i * DOUT + o];
    float v = mean + acc;
    if (do_leaky) v = leaky_f(v);
    out[idx] = v;
}

__global__ void zero_kernel(float* __restrict__ p, int n)
{
    const int i = blockIdx.x * blockDim.x + threadIdx.x;
    if (i < n) p[i] = 0.f;
}

extern "C" void kernel_launch(void* const* d_in, const int* in_sizes, int n_in,
                              void* d_out, int out_size, void* d_ws, size_t ws_size,
                              hipStream_t stream)
{
    const float* x    = (const float*)d_in[0];
    const int*   ei   = (const int*)d_in[1];
    const float* ea   = (const float*)d_in[2];
    const float* w1_0 = (const float*)d_in[3];
    const float* b1_0 = (const float*)d_in[4];
    const float* w2_0 = (const float*)d_in[5];
    const float* b2_0 = (const float*)d_in[6];
    const float* rt_0 = (const float*)d_in[7];
    const float* bs_0 = (const float*)d_in[8];
    const float* w1_1 = (const float*)d_in[9];
    const float* b1_1 = (const float*)d_in[10];
    const float* w2_1 = (const float*)d_in[11];
    const float* b2_1 = (const float*)d_in[12];
    const float* rt_1 = (const float*)d_in[13];
    const float* bs_1 = (const float*)d_in[14];

    float* wsf = (float*)d_ws;
    float* agg = wsf;                         // N*16
    float* cnt = wsf + (size_t)NNODES * 16;   // N
    float* h1  = wsf + (size_t)NNODES * 17;   // N*16
    float* out = (float*)d_out;

    const int nodeElems = NNODES * DOUT;
    const int nodeGrid  = (nodeElems + 255) / 256;

    // layer 0
    zero_kernel<<<(NNODES * 17 + 255) / 256, 256, 0, stream>>>(agg, NNODES * 17);
    nnconv_edge_kernel<<<EDGE_WGS, BLOCK, 0, stream>>>(x, ei, ea, w1_0, b1_0, w2_0, b2_0,
                                                       agg, cnt, 1);
    nnconv_node_kernel<<<nodeGrid, 256, 0, stream>>>(agg, cnt, x, rt_0, bs_0, h1, 1);

    // layer 1
    zero_kernel<<<(NNODES * 16 + 255) / 256, 256, 0, stream>>>(agg, NNODES * 16);
    nnconv_edge_kernel<<<EDGE_WGS, BLOCK, 0, stream>>>(h1, ei, ea, w1_1, b1_1, w2_1, b2_1,
                                                       agg, cnt, 0);
    nnconv_node_kernel<<<nodeGrid, 256, 0, stream>>>(agg, cnt, h1, rt_1, bs_1, out, 0);
}
